// DSSKernel_31817117729291
// MI455X (gfx1250) — matrix operand order repaired and checked by reading, not compiled, not run on hardware
//
#include <hip/hip_runtime.h>
#include <hip/hip_bf16.h>

// DSS kernel for gfx1250 (MI455X), wave32 + WMMA.
//
// K[h,l] = Re( sum_n W'[h,n] * exp(z[h,n]*(l - pos_n*(L-1))) ),  l in [0,2048)
// Factor l = q*64 + r  =>  per-head real GEMM (32 x 128)*(128 x 64) with
// K-dim = [Re || Im] and the Im half of B negated. Tiles computed with
// v_wmma_f32_16x16x32_f16 (f32 accumulate), operands staged in LDS already
// in WMMA fragment layout so each lane loads its fragment as 32 contiguous
// bytes (ds_load_b128).

typedef __attribute__((ext_vector_type(16))) _Float16 v16h;
typedef __attribute__((ext_vector_type(8)))  float    v8f;

constexpr int H_DIM = 512;
constexpr int N_DIM = 64;
constexpr int L_DIM = 2048;
constexpr int QT    = 32;    // l = q*RT + r
constexpr int RT    = 64;
constexpr float B_SCALE = 1024.0f;  // keep f16 B operands in a healthy range

__global__ __launch_bounds__(256)
void dss_wmma_kernel(const float* __restrict__ log_dt,   // (H,2)
                     const float* __restrict__ Lambda,   // (N,2)
                     const float* __restrict__ W,        // (1,H,N,2)
                     const int*   __restrict__ Lp,       // scalar L (=2048)
                     float* __restrict__ out)            // (1,H,L)
{
    const int h   = blockIdx.x;
    const int tid = threadIdx.x;
    const float Lf = (float)(*Lp);   // math uses L; tiling assumes L = QT*RT = 2048

    __shared__ float2 zsh[N_DIM];    // dt * Lambda (complex, split re/im dt)
    __shared__ float2 wsh[N_DIM];    // W * num * recip, pre-scaled by B_SCALE
    __shared__ float  psh[N_DIM];    // pos = (Lam.re > 0)
    // Operands stored directly in v_wmma_f32_16x16x32_f16 fragment layout:
    //   [m-tile][k-step][lane][element]  -> lane reads 16 halves contiguously.
    __shared__ _Float16 Afrag[2][4][32][16];   //  8 KB : A (32 x 128)
    __shared__ _Float16 Bfrag[4][4][32][16];   // 16 KB : B (128 x 64)

    // --- fragment-layout scatter helpers (inverse of ISA 7.12.2 tables) ---
    // A 16x32 f16: lane g*16+m holds K in {g*8+0..7, g*8+16..23}; element
    //   e<8 -> K = kbase+e (e==K&7 for K<8|16..23 split), per table.
    auto A_put = [&](int m, int k, _Float16 v) {
        int mt = m >> 4, lm = m & 15;
        int kk = k >> 5, kr = k & 31;
        int g, e;
        if      (kr < 8)  { g = 0; e = kr;      }
        else if (kr < 16) { g = 1; e = kr - 8;  }
        else if (kr < 24) { g = 0; e = kr - 8;  }
        else              { g = 1; e = kr - 16; }
        Afrag[mt][kk][g * 16 + lm][e] = v;
    };
    // B 32x16 f16 (same K order as A):
    //   lanes 0-15 hold K=kk*32+{0..7,16..23}, lanes 16-31 hold K=kk*32+{8..15,24..31}.
    auto B_put = [&](int k, int r, _Float16 v) {
        int nt = r >> 4, ln = r & 15;
        int kk = k >> 5, kr = k & 31;
        int g = (kr >> 3) & 1, e = (kr & 7) | ((kr >> 4) << 3);
        Bfrag[nt][kk][g * 16 + ln][e] = v;
    };

    // ---------------- phase 1: per-mode coefficients (64 threads) ----------
    if (tid < N_DIM) {
        const int n = tid;
        const float lre = Lambda[2 * n], lim = Lambda[2 * n + 1];
        const float dtr = expf(log_dt[2 * h]);
        const float dti = expf(log_dt[2 * h + 1]);
        const float zr  = dtr * lre, zi = dti * lim;
        const float pos = (lre > 0.0f) ? 1.0f : 0.0f;
        const float sgn = 1.0f - 2.0f * pos;
        const float znr = zr * sgn, zni = zi * sgn;

        float s1, c1, s2, c2;
        const float er = expf(znr);          sincosf(zni, &s1, &c1);
        const float el = expf(znr * Lf);     sincosf(zni * Lf, &s2, &c2);
        const float numr = er * c1 - 1.0f, numi = er * s1;   // exp(zn)-1
        const float denr = el * c2 - 1.0f, deni = el * s2;   // exp(zn*L)-1
        // x = den * Lam ; recip = conj(x)/(|x|^2 + eps)
        const float xr  = denr * lre - deni * lim;
        const float xi  = denr * lim + deni * lre;
        const float inv = 1.0f / (xr * xr + xi * xi + 1e-7f);
        const float rr  = xr * inv, ri = -xi * inv;
        // W' = Wc * num * recip
        const float wr  = W[(h * N_DIM + n) * 2];
        const float wi  = W[(h * N_DIM + n) * 2 + 1];
        const float t1r = wr * numr - wi * numi;
        const float t1i = wr * numi + wi * numr;
        const float wpr = t1r * rr - t1i * ri;
        const float wpi = t1r * ri + t1i * rr;

        zsh[n] = make_float2(zr, zi);
        wsh[n] = make_float2(wpr * B_SCALE, wpi * B_SCALE);
        psh[n] = pos;
    }
    __syncthreads();

    // ------------- phase 2a: A[q,n] = exp(z_n * 64 * (q - pos*31)) ---------
    for (int idx = tid; idx < QT * N_DIM; idx += 256) {
        const int q = idx >> 6, n = idx & 63;
        const float2 z = zsh[n];
        const float  t = (float)RT * ((float)q - psh[n] * (float)(QT - 1));
        float s, c;
        const float rad = expf(z.x * t);
        sincosf(z.y * t, &s, &c);
        A_put(q, n,          (_Float16)(c * rad));   // Re -> K row n
        A_put(q, n + N_DIM,  (_Float16)(s * rad));   // Im -> K row n+64
    }
    // --- phase 2b: B[n,r] = W'_n * exp(z_n * (r - pos*63)) (Im negated) ----
    for (int idx = tid; idx < N_DIM * RT; idx += 256) {
        const int n = idx >> 6, r = idx & 63;
        const float2 z = zsh[n];
        const float2 w = wsh[n];
        const float  t = (float)r - psh[n] * (float)(RT - 1);
        float s, c;
        const float rad = expf(z.x * t);
        sincosf(z.y * t, &s, &c);
        const float Sr = c * rad, Si = s * rad;
        const float vr = w.x * Sr - w.y * Si;
        const float vi = w.x * Si + w.y * Sr;
        B_put(n,         r, (_Float16)vr);
        B_put(n + N_DIM, r, (_Float16)(-vi));   // Re(A*B) = Ar*Br - Ai*Bi
    }
    __syncthreads();

    // ---------------- phase 3: one 16x16 output tile per wave --------------
    const int wave = tid >> 5, lane = tid & 31;
    const int mt = wave >> 2, nt = wave & 3;     // 2 x 4 tiles

    v8f acc = {};
#pragma unroll
    for (int kk = 0; kk < 4; ++kk) {
        v16h a = *(const v16h*)(&Afrag[mt][kk][lane][0]);   // 32B contiguous
        v16h b = *(const v16h*)(&Bfrag[nt][kk][lane][0]);
        acc = __builtin_amdgcn_wmma_f32_16x16x32_f16(
            /*neg_a=*/false, a, /*neg_b=*/false, b,
            /*c_mod=*/(short)0, acc, /*reuse_a=*/false, /*reuse_b=*/false);
    }

    // C/D layout: VGPR j -> (M = j + 8*(lane>=16), N = lane&15)
    const float invs  = 1.0f / B_SCALE;
    const int   rcol  = nt * 16 + (lane & 15);
    const int   qbase = mt * 16 + ((lane >> 4) << 3);
    float* outh = out + (size_t)h * L_DIM;
#pragma unroll
    for (int j = 0; j < 8; ++j) {
        const int q = qbase + j;
        outh[q * RT + rcol] = acc[j] * invs;
    }
}

extern "C" void kernel_launch(void* const* d_in, const int* in_sizes, int n_in,
                              void* d_out, int out_size, void* d_ws, size_t ws_size,
                              hipStream_t stream) {
    const float* log_dt = (const float*)d_in[0];   // (H,2)
    const float* Lambda = (const float*)d_in[1];   // (N,2)
    const float* W      = (const float*)d_in[2];   // (1,H,N,2)
    const int*   Lp     = (const int*)  d_in[3];   // scalar L
    float* out = (float*)d_out;                    // (1,H,L)

    dss_wmma_kernel<<<H_DIM, 256, 0, stream>>>(log_dt, Lambda, W, Lp, out);
}
